// SumSampler_91233695302291
// MI455X (gfx1250) — matrix operand order repaired and checked by reading, not compiled, not run on hardware
//
#include <hip/hip_runtime.h>

// SumSampler for MI455X (gfx1250, wave32).
// One block per batch row (512 blocks x 256 threads). All per-row state
// (probs table, 10^4-bin histogram, per-sample sums) lives in LDS (~44KB/WGP),
// so the histogram/gather phase needs no global atomics. Digit recombination
// d0*1000+d1*100+d2*10+d3 is done with v_wmma_f32_16x16x32_f16 (exact: digits
// and multipliers are exact f16, accumulation is f32) arranged so each lane
// reads back its own sample's value from D[0] with zero cross-lane shuffles.

#define NDIGITS  4
#define NSAMPLES 2048
#define NBATCH   512
#define NCLASS   10
#define NVALS    10000
#define TPB      256
#define SPT      (NSAMPLES / TPB)

typedef __attribute__((ext_vector_type(16))) _Float16 v16h;
typedef __attribute__((ext_vector_type(8)))  float    v8f;

__device__ __forceinline__ unsigned rotl32(unsigned x, int n) {
  return (x << n) | (x >> (32 - n));
}

// JAX threefry2x32 primitive: 5 rounds of 4 mixes + key injections.
__device__ __forceinline__ void threefry2x32(unsigned ks0, unsigned ks1,
                                             unsigned x0, unsigned x1,
                                             unsigned& o0, unsigned& o1) {
  const unsigned ks2 = ks0 ^ ks1 ^ 0x1BD11BDAu;
  x0 += ks0; x1 += ks1;
#define TF_MIX(r) { x0 += x1; x1 = rotl32(x1, (r)); x1 ^= x0; }
  TF_MIX(13) TF_MIX(15) TF_MIX(26) TF_MIX(6)
  x0 += ks1; x1 += ks2 + 1u;
  TF_MIX(17) TF_MIX(29) TF_MIX(16) TF_MIX(24)
  x0 += ks2; x1 += ks0 + 2u;
  TF_MIX(13) TF_MIX(15) TF_MIX(26) TF_MIX(6)
  x0 += ks0; x1 += ks1 + 3u;
  TF_MIX(17) TF_MIX(29) TF_MIX(16) TF_MIX(24)
  x0 += ks1; x1 += ks2 + 4u;
  TF_MIX(13) TF_MIX(15) TF_MIX(26) TF_MIX(6)
  x0 += ks2; x1 += ks0 + 5u;
#undef TF_MIX
  o0 = x0; o1 = x1;
}

// JAX gumbel from raw bits: u = max(tiny, f*(1-tiny)+tiny), f = bits->[0,1),
// g = -log(-log(u)).
__device__ __forceinline__ float gumbel_from_bits(unsigned bits) {
  const float TINY = 1.17549435e-38f;
  float f = __uint_as_float((bits >> 9) | 0x3f800000u) - 1.0f;
  float u = fmaxf(TINY, f * (1.0f - TINY) + TINY);
  return -logf(-logf(u));
}

__global__ __launch_bounds__(TPB) void SumSampler_kernel(
    const float* __restrict__ n1_logits,  // [B, D, 10]
    const float* __restrict__ counters,   // [D, 10]
    const int*   __restrict__ s,          // [B]
    float* __restrict__ out)              // n1[B,S,D] | n2d[B,S,D] | w[B,S]
{
  __shared__ float          probs_s[NDIGITS * NCLASS]; // logits - 0.1*log(cnt)
  __shared__ unsigned       hist_s[NVALS];
  __shared__ unsigned short summed_s[NSAMPLES];

  const int b = blockIdx.x;
  const int t = threadIdx.x;
  const unsigned lane = (unsigned)(t & 31);

  __builtin_prefetch(n1_logits + (size_t)b * NDIGITS * NCLASS, 0, 0);

  for (int h = t; h < NVALS; h += TPB) hist_s[h] = 0u;
  if (t < NDIGITS * NCLASS) {
    probs_s[t] = n1_logits[(size_t)b * NDIGITS * NCLASS + t]
               - 0.1f * logf(counters[t]);
  }
  __syncthreads();

  // Subkey chain: key=(0,42); split(): new=(y0(0,2),y0(1,3)), sub=(y1(0,2),y1(1,3)).
  unsigned sk0[NDIGITS], sk1[NDIGITS];
  {
    unsigned k0 = 0u, k1 = 42u;
#pragma unroll
    for (int i = 0; i < NDIGITS; ++i) {
      unsigned a0, a1, c0, c1;
      threefry2x32(k0, k1, 0u, 2u, a0, a1);
      threefry2x32(k0, k1, 1u, 3u, c0, c1);
      sk0[i] = a1; sk1[i] = c1;   // sub key for digit i
      k0 = a0;     k1 = c0;       // carried key
    }
  }

  // A matrix (f16 16x32): rows 0-7 hold multipliers at K=0..3, rows 8-15 at
  // K=8..11. Lane m<8: elements 0..3; lanes 24..31 (rows 8-15, h=1): elements
  // 0..3. All other lanes/elements -> zero.
  v16h aM = {};
  if (lane < 8u || lane >= 24u) {
    aM[0] = (_Float16)1000.0f; aM[1] = (_Float16)100.0f;
    aM[2] = (_Float16)10.0f;   aM[3] = (_Float16)1.0f;
  }

  const int sv = s[b];
  const int m0 = sv / 1000;
  const int m1 = (sv / 100) % 10;
  const int m2 = (sv / 10) % 10;
  const int m3 = sv % 10;
  const int maxims[NDIGITS] = { m0, m1, m2, m3 };
  const int min0 = (m0 - 9 > 0) ? (m0 - 9) : 0;   // always 0 here, kept general

  const unsigned HALF = (unsigned)NBATCH * NSAMPLES * NCLASS / 2u;
  const size_t OFF1 = (size_t)NBATCH * NSAMPLES * NDIGITS;
  const size_t OFF2 = 2u * OFF1;

  for (int j = 0; j < SPT; ++j) {
    const int sIdx = t + j * TPB;
    const unsigned row = (unsigned)(b * NSAMPLES + sIdx);
    const unsigned base = row * (unsigned)NCLASS;
    const bool lo_half = base < HALF;   // whole class block lies in one half

    int constraint = 0;
    float dig[NDIGITS];

#pragma unroll
    for (int i = 0; i < NDIGITS; ++i) {
      const int maxim = maxims[i];
      const int minim = (i == 0) ? min0 : 0;
      const unsigned k0 = sk0[i], k1 = sk1[i];
      float best = -__builtin_inff();
      int bestc = 0;
      for (int c = 0; c < NCLASS; ++c) {
        const unsigned idx = base + (unsigned)c;
        unsigned o0, o1, bits;
        if (lo_half) { threefry2x32(k0, k1, idx, idx + HALF, o0, o1); bits = o0; }
        else         { threefry2x32(k0, k1, idx - HALF, idx, o0, o1); bits = o1; }
        const float g = gumbel_from_bits(bits);
        const bool valid = (c >= minim) && (c <= maxim);
        const float lg = ((constraint != 0) || valid)
                             ? (probs_s[i * NCLASS + c] + g)
                             : -__builtin_inff();
        if (lg > best) { best = lg; bestc = c; }   // first-occurrence argmax
      }
      dig[i] = (float)bestc;
      constraint |= (bestc != maxim) ? 1 : 0;
    }

    // n1 samples output (float4 per lane, contiguous, 16B aligned)
    reinterpret_cast<float4*>(out)[row] =
        make_float4(dig[0], dig[1], dig[2], dig[3]);

    // Recombine digits via WMMA: B elements 0..3 = own digits. Lanes 0-15 feed
    // K=0..3 (cols 0-15), lanes 16-31 feed K=8..11. D[0] at lane L = value of
    // lane L's own sample.
    v16h bM = {};
    bM[0] = (_Float16)dig[0]; bM[1] = (_Float16)dig[1];
    bM[2] = (_Float16)dig[2]; bM[3] = (_Float16)dig[3];
    v8f cz = {};
    v8f dM = __builtin_amdgcn_wmma_f32_16x16x32_f16(
        false, aM, false, bM, (short)0, cz, false, false);
    const int n1s = (int)(dM[0] + 0.25f);   // exact integer in f32

    const int n2 = sv - n1s;                // guaranteed in [0, 10^4)
    const int q0 = n2 / 1000;  const int r0 = n2 - q0 * 1000;
    const int q1 = r0 / 100;   const int r1 = r0 - q1 * 100;
    const int q2 = r1 / 10;    const int q3 = r1 - q2 * 10;
    reinterpret_cast<float4*>(out + OFF1)[row] =
        make_float4((float)q0, (float)q1, (float)q2, (float)q3);

    summed_s[sIdx] = (unsigned short)n1s;
    atomicAdd(&hist_s[n1s], 1u);
  }
  __syncthreads();

  for (int j = 0; j < SPT; ++j) {
    const int sIdx = t + j * TPB;
    const unsigned w = hist_s[summed_s[sIdx]];
    out[OFF2 + (size_t)b * NSAMPLES + sIdx] = (float)w;
  }
}

extern "C" void kernel_launch(void* const* d_in, const int* in_sizes, int n_in,
                              void* d_out, int out_size, void* d_ws, size_t ws_size,
                              hipStream_t stream) {
  (void)in_sizes; (void)n_in; (void)out_size; (void)d_ws; (void)ws_size;
  const float* n1_logits = (const float*)d_in[0];
  const float* counters  = (const float*)d_in[1];
  const int*   sv        = (const int*)d_in[2];
  float* out = (float*)d_out;
  SumSampler_kernel<<<dim3(NBATCH), dim3(TPB), 0, stream>>>(
      n1_logits, counters, sv, out);
}
